// EnhancedContextAwareDualVQ_24902220382527
// MI455X (gfx1250) — matrix operand order repaired and checked by reading, not compiled, not run on hardware
//
#include <hip/hip_runtime.h>
#include <hip/hip_bf16.h>

typedef __attribute__((ext_vector_type(16))) __bf16 v16bf;
typedef __attribute__((ext_vector_type(8)))  float  v8f;
typedef __attribute__((ext_vector_type(4)))  float  v4f;

#define DIM        256
#define N_EMB      1024
#define N_ROWS     32768      // B*L = 8*4096
#define KCHUNKS    8          // 256 / 32
#define NTILES     64         // 1024 / 16
#define ROWS_PER_WG 128       // 8 waves * 16 rows
#define EBF_BYTES  (N_EMB * DIM * 2)   // 512 KB bf16 codebook, B-fragment layout
#define OUT_ELEMS  (N_ROWS * DIM)      // 8388608 per output tensor

// ---------------------------------------------------------------------------
// Pack codebook into WMMA B-fragment layout (bf16).
// B tile for N-tile t, K-chunk kc is 32x16 (KxN): lane l holds column
// n = t*16 + (l%16), K = kc*32 + 16*(l/16) + [0..15]  contiguously -> v16bf.
// ---------------------------------------------------------------------------
__global__ void prep_ebf(const float* __restrict__ emb, __bf16* __restrict__ ebf) {
    const int t    = blockIdx.x;           // 0..63  N-tile
    const int kc   = threadIdx.x >> 5;     // 0..7   K-chunk
    const int lane = threadIdx.x & 31;
    const int n     = t * 16 + (lane & 15);
    const int kbase = kc * 32 + 16 * (lane >> 4);
    const float* __restrict__ src = emb + n * DIM + kbase;
    v16bf v;
#pragma unroll
    for (int e = 0; e < 16; ++e) v[e] = (__bf16)src[e];
    v16bf* dst = (v16bf*)(ebf + (((size_t)(t * KCHUNKS + kc) * 32 + lane) * 16));
    *dst = v;
}

// ---------------------------------------------------------------------------
// Exact f32 codebook row norms; also zero the loss accumulator.
// ---------------------------------------------------------------------------
__global__ void prep_enorm(const float* __restrict__ emb, float* __restrict__ enorm,
                           float* __restrict__ loss) {
    const int n = blockIdx.x * blockDim.x + threadIdx.x;
    if (n == 0) *loss = 0.0f;
    if (n < N_EMB) {
        const v4f* __restrict__ p = (const v4f*)(emb + n * DIM);
        float s = 0.0f;
#pragma unroll 8
        for (int i = 0; i < DIM / 4; ++i) {
            v4f x = p[i];
            s = __builtin_fmaf(x.x, x.x, s);
            s = __builtin_fmaf(x.y, x.y, s);
            s = __builtin_fmaf(x.z, x.z, s);
            s = __builtin_fmaf(x.w, x.w, s);
        }
        enorm[n] = s;
    }
}

// ---------------------------------------------------------------------------
// Main VQ kernel: 8 waves/block, each wave owns 16 rows, scans 64 codebook
// tiles with v_wmma_f32_16x16x32_bf16, fused argmin of ||e||^2 - 2 z.e,
// then exact-f32 gather + pass-through + loss.
// ---------------------------------------------------------------------------
__global__ __launch_bounds__(256) void vq_main(
    const float*  __restrict__ z_real,
    const float*  __restrict__ z_imag,
    const float*  __restrict__ emb,
    const __bf16* __restrict__ ebf,
    const float*  __restrict__ enorm,
    float* __restrict__ out_real,
    float* __restrict__ out_imag,
    float* __restrict__ loss)
{
    const int lane = threadIdx.x & 31;
    const int wave = threadIdx.x >> 5;
    const int m    = lane & 15;        // row within tile (A) / column n offset (B,C)
    const int h    = lane >> 4;        // lane half
    const int row_base = blockIdx.x * ROWS_PER_WG + wave * 16;

    // ---- Build A fragments (16x32 bf16 per K-chunk) from this wave's 16 rows.
    // Layout (matches packed B): lane(m,h) holds K = kc*32 + 16h + [0..15].
    const float* __restrict__ rowptr = z_real + (size_t)(row_base + m) * DIM;
    v16bf a[KCHUNKS];
#pragma unroll
    for (int kc = 0; kc < KCHUNKS; ++kc) {
        const int k0 = kc * 32 + 16 * h;
#pragma unroll
        for (int j = 0; j < 8; ++j) a[kc][j]     = (__bf16)rowptr[k0 + j];
#pragma unroll
        for (int j = 0; j < 8; ++j) a[kc][8 + j] = (__bf16)rowptr[k0 + 8 + j];
    }

    // ---- Running per-lane argmin (8 C-rows per lane, column n = t*16 + m).
    float rmin[8];
    int   ridx[8];
#pragma unroll
    for (int c = 0; c < 8; ++c) { rmin[c] = 3.4e38f; ridx[c] = 0; }

    const __bf16* __restrict__ bb = ebf + (size_t)lane * 16;
    for (int t = 0; t < NTILES; ++t) {
        const __bf16* bt = bb + (size_t)t * (KCHUNKS * 32 * 16);
        if (t + 1 < NTILES)
            __builtin_prefetch(bt + (size_t)(KCHUNKS * 32 * 16), 0, 1);  // global_prefetch_b8
        v8f acc = {};
#pragma unroll
        for (int kc = 0; kc < KCHUNKS; ++kc) {
            v16bf b = *(const v16bf*)(bt + kc * (32 * 16));
            acc = __builtin_amdgcn_wmma_f32_16x16x32_bf16(
                      false, a[kc], false, b, (short)0, acc, false, false);
        }
        const float en = enorm[t * 16 + m];
        const int   n  = t * 16 + m;
#pragma unroll
        for (int c = 0; c < 8; ++c) {
            float d = __builtin_fmaf(-2.0f, acc[c], en);   // ||e||^2 - 2 z.e
            if (d < rmin[c]) { rmin[c] = d; ridx[c] = n; }
        }
    }

    // ---- Reduce argmin over the 16 columns inside each lane-half.
    // Masks 1..8 never cross the 16-lane boundary in wave32.
#pragma unroll
    for (int c = 0; c < 8; ++c) {
#pragma unroll
        for (int off = 8; off >= 1; off >>= 1) {
            float om = __shfl_xor(rmin[c], off, 32);
            int   oi = __shfl_xor(ridx[c], off, 32);
            if (om < rmin[c] || (om == rmin[c] && oi < ridx[c])) {
                rmin[c] = om; ridx[c] = oi;
            }
        }
    }

    // Broadcast the winning index for each of this wave's 16 rows to all lanes.
    // Rows 0..7 live in lanes 0..15 (c = row), rows 8..15 in lanes 16..31.
    int rowidx[16];
#pragma unroll
    for (int r = 0; r < 8; ++r) rowidx[r]     = __shfl(ridx[r], 0, 32);
#pragma unroll
    for (int r = 0; r < 8; ++r) rowidx[8 + r] = __shfl(ridx[r], 16, 32);

    // ---- Output stage: exact f32 gather, imag pass-through, loss partial.
    float ss = 0.0f;
#pragma unroll
    for (int r = 0; r < 16; ++r) {
        const size_t grow = (size_t)(row_base + r) * DIM + lane * 8;
        const float* __restrict__ ep = emb + (size_t)rowidx[r] * DIM + lane * 8;
        const float* __restrict__ zp = z_real + grow;
        const float* __restrict__ zi = z_imag + grow;
        float* __restrict__ orp = out_real + grow;
        float* __restrict__ oip = out_imag + grow;
#pragma unroll
        for (int j = 0; j < 8; ++j) {
            const float e = ep[j];
            const float d = e - zp[j];
            ss = __builtin_fmaf(d, d, ss);
            orp[j] = e;        // z + stop(zq - z) == zq (forward value)
            oip[j] = zi[j];    // imag passes through
        }
    }
#pragma unroll
    for (int off = 16; off >= 1; off >>= 1)
        ss += __shfl_xor(ss, off, 32);
    if (lane == 0)
        atomicAdd(loss, ss * (1.25f / (float)OUT_ELEMS));  // mse + 0.25*mse
}

// ---------------------------------------------------------------------------
extern "C" void kernel_launch(void* const* d_in, const int* in_sizes, int n_in,
                              void* d_out, int out_size, void* d_ws, size_t ws_size,
                              hipStream_t stream) {
    const float* z_real = (const float*)d_in[0];
    const float* z_imag = (const float*)d_in[1];
    const float* emb    = (const float*)d_in[2];

    float* out      = (float*)d_out;
    float* out_real = out;
    float* out_imag = out + OUT_ELEMS;
    float* loss     = out + 2 * (size_t)OUT_ELEMS;

    __bf16* ebf   = (__bf16*)d_ws;
    float*  enorm = (float*)((char*)d_ws + EBF_BYTES);

    prep_ebf<<<NTILES, 256, 0, stream>>>(emb, ebf);
    prep_enorm<<<(N_EMB + 255) / 256, 256, 0, stream>>>(emb, enorm, loss);
    vq_main<<<N_ROWS / ROWS_PER_WG, 256, 0, stream>>>(
        z_real, z_imag, emb, ebf, enorm, out_real, out_imag, loss);
}
